// MultiheadAttention_51247549776137
// MI455X (gfx1250) — matrix operand order repaired and checked by reading, not compiled, not run on hardware
//
#include <hip/hip_runtime.h>

namespace {

constexpr int T_SEQ  = 1024;
constexpr int BATCH  = 4;
constexpr int DMODEL = 1024;
constexpr int NHEAD  = 16;
constexpr int DHEAD  = 64;
constexpr int TB     = T_SEQ * BATCH;   // 4096 rows (t*B + b)
constexpr int HDIM   = NHEAD * DHEAD;   // 1024
constexpr float SCORE_SCALE = 0.125f;   // 1/sqrt(64)
constexpr float NEG_INF = -3.0e38f;

typedef __attribute__((ext_vector_type(16))) __bf16        v16bf;
typedef __attribute__((ext_vector_type(8)))  float         v8f;
typedef __attribute__((ext_vector_type(4)))  unsigned int  u32x4;
typedef __attribute__((ext_vector_type(8)))  int           i32x8;
typedef __attribute__((ext_vector_type(4)))  int           i32x4;
typedef __attribute__((ext_vector_type(4)))  float         f32x4;
typedef unsigned short u16;

union Frag16 { u32x4 u[2]; v16bf v; };
union Acc8   { v8f v; float f[8]; };

__device__ __forceinline__ u16 f32_to_bf16(float x) {
  unsigned int u = __float_as_uint(x);
  u += 0x7FFFu + ((u >> 16) & 1u);          // round-to-nearest-even
  return (u16)(u >> 16);
}

// A fragment (16x32 bf16, M x K): lanes 0-15 row M=lane hold K {k0..k0+7, k0+16..k0+23},
// lanes 16-31 row M=lane-16 hold K {k0+8.., k0+24..}. Two 16B loads per lane.
__device__ __forceinline__ v16bf load_fragA(const u16* __restrict__ base, int ld,
                                            int rowBase, int k0, int lane) {
  const int r  = rowBase + (lane & 15);
  const int kk = k0 + ((lane >> 4) << 3);
  const u16* p = base + (size_t)r * ld + kk;
  Frag16 f;
  f.u[0] = *(const u32x4*)(p);
  f.u[1] = *(const u32x4*)(p + 16);
  return f.v;
}

// B fragment (32x16 bf16, K x N) from transposed storage (N-major, K contiguous):
// lane n = column, lanes 0-15 hold K {k0..k0+7, k0+16..k0+23}, lanes 16-31 hold K {k0+8.., k0+24..}.
__device__ __forceinline__ v16bf load_fragB(const u16* __restrict__ baseT, int ld,
                                            int colBase, int k0, int lane) {
  const int n  = colBase + (lane & 15);
  const int kk = k0 + ((lane >> 4) << 3);
  const u16* p = baseT + (size_t)n * ld + kk;
  Frag16 f;
  f.u[0] = *(const u32x4*)(p);
  f.u[1] = *(const u32x4*)(p + 16);
  return f.v;
}

// B fragment from an LDS-resident 64x32 tile laid out [n_local][k] (TDM destination).
__device__ __forceinline__ v16bf load_fragB_lds(const u16* lbase, int colBase, int lane) {
  const int n  = colBase + (lane & 15);
  const int kk = (lane >> 4) << 3;
  const u16* p = lbase + n * 32 + kk;
  Frag16 f;
  f.u[0] = *(const u32x4*)(p);
  f.u[1] = *(const u32x4*)(p + 16);
  return f.v;
}

__device__ __forceinline__ v8f wmma_bf16(v16bf a, v16bf b, v8f c) {
  return __builtin_amdgcn_wmma_f32_16x16x32_bf16(false, a, false, b, (short)0, c,
                                                 false, false);
}

// ---- Tensor Data Mover: DMA a 64-row x 32-elem bf16 tile (rows strided by 1024
// elements) from global memory into LDS at lds_off. 2-D descriptor, groups 2/3 null.
__device__ __forceinline__ void tdm_load_tile_64x32(const u16* gptr, unsigned lds_off) {
  const unsigned long long ga = (unsigned long long)(size_t)gptr;
  u32x4 g0;
  g0[0] = 1u;                                    // count=1, user descriptor
  g0[1] = lds_off;                               // lds_addr (bytes)
  g0[2] = (unsigned)ga;                          // global_addr[31:0]
  g0[3] = (unsigned)((ga >> 32) & 0x1FFFFFFu)    // global_addr[56:32]
        | (2u << 30);                            // type = 2 ("image")
  i32x8 g1;
  g1[0] = 0x00010000;        // workgroup_mask=0, data_size=1 (2 bytes)
  g1[1] = 0x04000000;        // tensor_dim0 = 1024 (low 16 in [31:16])
  g1[2] = 0x04000000;        // tensor_dim0 hi=0, tensor_dim1 = 1024 (low 16 in [31:16])
  g1[3] = 0x00200000;        // tensor_dim1 hi=0, tile_dim0 = 32 elements
  g1[4] = 64;                // tile_dim1 = 64 rows, tile_dim2 = 0
  g1[5] = 1024;              // tensor_dim0_stride = 1024 elements (low 32)
  g1[6] = 0;                 // stride0 hi, tensor_dim1_stride low = 0
  g1[7] = 0;
  i32x4 g2 = {0, 0, 0, 0};
  i32x4 g3 = {0, 0, 0, 0};
#if __clang_major__ >= 23
  i32x8 g4 = {0, 0, 0, 0, 0, 0, 0, 0};
  __builtin_amdgcn_tensor_load_to_lds(g0, g1, g2, g3, g4, 0);
#else
  __builtin_amdgcn_tensor_load_to_lds(g0, g1, g2, g3, 0);
#endif
}

// ---------------- elementwise conversion f32 -> bf16 ----------------
__global__ __launch_bounds__(256)
void cvt_f32_to_bf16(const float* __restrict__ in, u16* __restrict__ out, int n) {
  int idx = (blockIdx.x * 256 + threadIdx.x) * 4;
  if (idx + 3 < n) {
    f32x4 v = *(const f32x4*)(in + idx);
    out[idx + 0] = f32_to_bf16(v[0]);
    out[idx + 1] = f32_to_bf16(v[1]);
    out[idx + 2] = f32_to_bf16(v[2]);
    out[idx + 3] = f32_to_bf16(v[3]);
  }
}

// ------------- transpose + convert: (R,C) f32 -> (C,R) bf16 -------------
__global__ __launch_bounds__(256)
void transpose_to_bf16(const float* __restrict__ in, u16* __restrict__ out,
                       int R, int C) {
  int idx = blockIdx.x * 256 + threadIdx.x;
  if (idx < R * C) {
    int r = idx / C, c = idx - r * C;
    out[(size_t)c * R + r] = f32_to_bf16(in[idx]);
  }
}

// ---------------- bf16 WMMA GEMM: (4096 x 1024) = A(4096x1024) * Bt^T ----------------
// B tile (64 x 32 per workgroup per k-step) is staged into LDS by the Tensor Data
// Mover, double-buffered, issued by wave 0 and synchronized via TENSORcnt + barrier.
// MODE 0: store bf16 to (B,NHEAD,T,DHEAD)   (hq / hk)
// MODE 1: store bf16 to (B,NHEAD,DHEAD,T)   (hv transposed)
// MODE 2: store f32  to row-major (m, n)    (attn_out)
template <int MODE>
__global__ __launch_bounds__(256)
void gemm4096x1024_bf16(const u16* __restrict__ A, const u16* __restrict__ Bt,
                        void* __restrict__ outp) {
  __shared__ __align__(16) u16 bTile[2][64 * 32];   // 2 x 4KB double buffer

  const int lane  = threadIdx.x & 31;
  const int w     = threadIdx.x >> 5;
  const int mBase = blockIdx.y * 128 + (w & 3) * 32;
  const int nLoc  = (w >> 2) * 32;                  // local column base in tile
  const int nBase = blockIdx.x * 64 + nLoc;

  const u16* BtWG = Bt + (size_t)(blockIdx.x * 64) * DMODEL;

  Acc8 acc[2][2];
#pragma unroll
  for (int mi = 0; mi < 2; mi++)
#pragma unroll
    for (int ni = 0; ni < 2; ni++)
#pragma unroll
      for (int i = 0; i < 8; i++) acc[mi][ni].f[i] = 0.0f;

  if (w == 0)   // one TDM issue per workgroup
    tdm_load_tile_64x32(BtWG, (unsigned)(size_t)&bTile[0][0]);

  for (int k0 = 0; k0 < DMODEL; k0 += 32) {
    const int buf = (k0 >> 5) & 1;
    if (w == 0) {
      if (k0 + 32 < DMODEL) {
        tdm_load_tile_64x32(BtWG + k0 + 32, (unsigned)(size_t)&bTile[buf ^ 1][0]);
        __builtin_amdgcn_s_wait_tensorcnt(1);   // current tile landed (in-order TDM)
      } else {
        __builtin_amdgcn_s_wait_tensorcnt(0);
      }
    }
    __syncthreads();   // publish LDS tile to all waves

    if (k0 + 32 < DMODEL)
      __builtin_prefetch(A + (size_t)(mBase + (lane & 15)) * DMODEL + k0 + 32, 0, 1);

    v16bf a0 = load_fragA(A, DMODEL, mBase,      k0, lane);
    v16bf a1 = load_fragA(A, DMODEL, mBase + 16, k0, lane);
    v16bf b0 = load_fragB_lds(&bTile[buf][0], nLoc,      lane);
    v16bf b1 = load_fragB_lds(&bTile[buf][0], nLoc + 16, lane);
    acc[0][0].v = wmma_bf16(a0, b0, acc[0][0].v);
    acc[0][1].v = wmma_bf16(a0, b1, acc[0][1].v);
    acc[1][0].v = wmma_bf16(a1, b0, acc[1][0].v);
    acc[1][1].v = wmma_bf16(a1, b1, acc[1][1].v);

    __syncthreads();   // protect buffer reused by next iteration's TDM write
  }

  const int hi = lane >> 4, nn = lane & 15;
#pragma unroll
  for (int mi = 0; mi < 2; mi++)
#pragma unroll
    for (int ni = 0; ni < 2; ni++)
#pragma unroll
      for (int i = 0; i < 8; i++) {
        int m = mBase + mi * 16 + i + 8 * hi;   // token index t*BATCH + b
        int n = nBase + ni * 16 + nn;           // output feature
        float val = acc[mi][ni].f[i];
        if (MODE == 2) {
          ((float*)outp)[(size_t)m * HDIM + n] = val;
        } else {
          int t = m >> 2, b = m & 3;
          int nh = n >> 6, d = n & 63;
          u16* ob = (u16*)outp;
          if (MODE == 0)
            ob[(((size_t)(b * NHEAD + nh)) * T_SEQ + t) * DHEAD + d] = f32_to_bf16(val);
          else
            ob[(((size_t)(b * NHEAD + nh)) * DHEAD + d) * T_SEQ + t] = f32_to_bf16(val);
        }
      }
}

// ---------------- fused causal flash attention, one wave per 16 query rows ----------------
__global__ __launch_bounds__(32)
void attn_fused(const u16* __restrict__ hq, const u16* __restrict__ hk,
                const u16* __restrict__ hvT, u16* __restrict__ attn_vec) {
  __shared__ __align__(16) u16 lds_p[16 * 32];   // 16 query rows x 32 key cols, bf16

  const int lane = threadIdx.x & 31;
  const int hi = lane >> 4, nn = lane & 15;
  const int bh = blockIdx.y;                 // b*NHEAD + nh
  const int q0 = blockIdx.x * 16;

  const u16* Q  = hq  + (size_t)bh * T_SEQ * DHEAD;
  const u16* Kb = hk  + (size_t)bh * T_SEQ * DHEAD;
  const u16* Vt = hvT + (size_t)bh * DHEAD * T_SEQ;

  v16bf qf0 = load_fragA(Q, DHEAD, q0, 0, lane);
  v16bf qf1 = load_fragA(Q, DHEAD, q0, 32, lane);

  Acc8 o[4];
  float mrow[8], lrow[8];
#pragma unroll
  for (int i = 0; i < 8; i++) { mrow[i] = NEG_INF; lrow[i] = 0.0f; }
#pragma unroll
  for (int ot = 0; ot < 4; ot++)
#pragma unroll
    for (int i = 0; i < 8; i++) o[ot].f[i] = 0.0f;

  for (int j0 = 0; j0 <= q0 + 15; j0 += 32) {
    // ---- S = Q * K^T for a 16 x 32 key block ----
    Acc8 s[2];
#pragma unroll
    for (int jt = 0; jt < 2; jt++) {
#pragma unroll
      for (int i = 0; i < 8; i++) s[jt].f[i] = 0.0f;
      v16bf kf0 = load_fragB(Kb, DHEAD, j0 + jt * 16, 0, lane);
      v16bf kf1 = load_fragB(Kb, DHEAD, j0 + jt * 16, 32, lane);
      s[jt].v = wmma_bf16(qf0, kf0, s[jt].v);
      s[jt].v = wmma_bf16(qf1, kf1, s[jt].v);
    }

    // ---- scale + causal mask + row max (rows live in 16-lane halves) ----
    float rmax[8];
#pragma unroll
    for (int i = 0; i < 8; i++) rmax[i] = NEG_INF;
#pragma unroll
    for (int jt = 0; jt < 2; jt++)
#pragma unroll
      for (int i = 0; i < 8; i++) {
        int qi = q0 + i + 8 * hi;
        int kj = j0 + jt * 16 + nn;
        float sv = s[jt].f[i] * SCORE_SCALE;
        sv = (kj > qi) ? NEG_INF : sv;
        s[jt].f[i] = sv;
        rmax[i] = fmaxf(rmax[i], sv);
      }
#pragma unroll
    for (int off = 1; off < 16; off <<= 1)
#pragma unroll
      for (int i = 0; i < 8; i++)
        rmax[i] = fmaxf(rmax[i], __shfl_xor(rmax[i], off, 32));

    // ---- online softmax: P = exp(S - m), store P (bf16) to LDS in C layout ----
    float mnew[8], corr[8], rsum[8];
#pragma unroll
    for (int i = 0; i < 8; i++) {
      mnew[i] = fmaxf(mrow[i], rmax[i]);
      corr[i] = __expf(mrow[i] - mnew[i]);
      rsum[i] = 0.0f;
    }
#pragma unroll
    for (int jt = 0; jt < 2; jt++)
#pragma unroll
      for (int i = 0; i < 8; i++) {
        float p = __expf(s[jt].f[i] - mnew[i]);
        rsum[i] += p;
        lds_p[(i + 8 * hi) * 32 + jt * 16 + nn] = f32_to_bf16(p);
      }
#pragma unroll
    for (int off = 1; off < 16; off <<= 1)
#pragma unroll
      for (int i = 0; i < 8; i++)
        rsum[i] += __shfl_xor(rsum[i], off, 32);
#pragma unroll
    for (int i = 0; i < 8; i++) {
      lrow[i] = lrow[i] * corr[i] + rsum[i];
      mrow[i] = mnew[i];
    }
#pragma unroll
    for (int ot = 0; ot < 4; ot++)
#pragma unroll
      for (int i = 0; i < 8; i++) o[ot].f[i] *= corr[i];

    __syncthreads();   // single-wave WG: barrier is a NOP, but forces s_wait_dscnt fence

    // ---- reload P as an A fragment (16x32) ----
    Frag16 pf;
    {
      const u16* lp = &lds_p[(lane & 15) * 32 + (hi << 3)];
      pf.u[0] = *(const u32x4*)(lp);
      pf.u[1] = *(const u32x4*)(lp + 16);
    }

    // ---- O += P * V_block : 4 N-tiles over d ----
#pragma unroll
    for (int ot = 0; ot < 4; ot++) {
      v16bf vf = load_fragB(Vt, T_SEQ, ot * 16, j0, lane);
      o[ot].v = wmma_bf16(pf.v, vf, o[ot].v);
    }
    __syncthreads();
  }

  // ---- normalize and store attn_vec as (T,B,NHEAD,DHEAD) bf16 ----
  const int b = bh >> 4, nh = bh & 15;
#pragma unroll
  for (int ot = 0; ot < 4; ot++)
#pragma unroll
    for (int i = 0; i < 8; i++) {
      float val = o[ot].f[i] / lrow[i];
      int t = q0 + i + 8 * hi;
      int d = ot * 16 + nn;
      attn_vec[(((size_t)t * BATCH + b) * NHEAD + nh) * DHEAD + d] = f32_to_bf16(val);
    }
}

// ---------------- row layernorm over d_model=1024, one block per row ----------------
__global__ __launch_bounds__(256)
void layernorm_rows(const float* __restrict__ x, const float* __restrict__ gamma,
                    const float* __restrict__ beta, float* __restrict__ out) {
  __shared__ float redS[8], redS2[8];
  const int row = blockIdx.x;
  const int tid = threadIdx.x;
  const float* xr = x + (size_t)row * DMODEL;

  f32x4 v = *(const f32x4*)(xr + tid * 4);
  float s  = v[0] + v[1] + v[2] + v[3];
  float s2 = v[0]*v[0] + v[1]*v[1] + v[2]*v[2] + v[3]*v[3];
#pragma unroll
  for (int off = 1; off < 32; off <<= 1) {
    s  += __shfl_xor(s,  off, 32);
    s2 += __shfl_xor(s2, off, 32);
  }
  if ((tid & 31) == 0) { redS[tid >> 5] = s; redS2[tid >> 5] = s2; }
  __syncthreads();
  s = 0.0f; s2 = 0.0f;
#pragma unroll
  for (int i = 0; i < 8; i++) { s += redS[i]; s2 += redS2[i]; }

  const float mu   = s * (1.0f / DMODEL);
  const float var  = s2 * (1.0f / DMODEL) - mu * mu;
  const float rstd = rsqrtf(var + 1e-3f);
#pragma unroll
  for (int i = 0; i < 4; i++) {
    int col = tid * 4 + i;
    out[(size_t)row * DMODEL + col] = (v[i] - mu) * rstd * gamma[col] + beta[col];
  }
}

} // anonymous namespace

extern "C" void kernel_launch(void* const* d_in, const int* in_sizes, int n_in,
                              void* d_out, int out_size, void* d_ws, size_t ws_size,
                              hipStream_t stream) {
  (void)in_sizes; (void)n_in; (void)out_size; (void)ws_size;

  const float* q     = (const float*)d_in[0];
  const float* k     = (const float*)d_in[1];
  const float* v     = (const float*)d_in[2];
  /* d_in[3] = attn_mask (bool, causal) — implemented analytically in-kernel */
  const float* Wq    = (const float*)d_in[4];
  const float* Wk    = (const float*)d_in[5];
  const float* Wv    = (const float*)d_in[6];
  const float* Wo    = (const float*)d_in[7];
  const float* gamma = (const float*)d_in[8];
  const float* beta  = (const float*)d_in[9];

  char* ws = (char*)d_ws;
  size_t off = 0;
  auto alloc = [&](size_t bytes) -> void* {
    void* p = ws + off;
    off += (bytes + 255) & ~(size_t)255;
    return p;
  };

  u16* qb   = (u16*)alloc((size_t)TB * DMODEL * 2);
  u16* kb   = (u16*)alloc((size_t)TB * DMODEL * 2);
  u16* vb   = (u16*)alloc((size_t)TB * DMODEL * 2);
  u16* WqT  = (u16*)alloc((size_t)DMODEL * HDIM * 2);
  u16* WkT  = (u16*)alloc((size_t)DMODEL * HDIM * 2);
  u16* WvT  = (u16*)alloc((size_t)DMODEL * HDIM * 2);
  u16* WoT  = (u16*)alloc((size_t)HDIM * DMODEL * 2);
  u16* hq   = (u16*)alloc((size_t)BATCH * NHEAD * T_SEQ * DHEAD * 2);
  u16* hk   = (u16*)alloc((size_t)BATCH * NHEAD * T_SEQ * DHEAD * 2);
  u16* hvT  = (u16*)alloc((size_t)BATCH * NHEAD * DHEAD * T_SEQ * 2);
  u16* avec = (u16*)alloc((size_t)TB * HDIM * 2);
  float* attn_out = (float*)alloc((size_t)TB * DMODEL * 4);

  const int nElem = TB * DMODEL;                // 4M
  cvt_f32_to_bf16<<<nElem / 1024, 256, 0, stream>>>(q, qb, nElem);
  cvt_f32_to_bf16<<<nElem / 1024, 256, 0, stream>>>(k, kb, nElem);
  cvt_f32_to_bf16<<<nElem / 1024, 256, 0, stream>>>(v, vb, nElem);

  const int wElem = DMODEL * HDIM;              // 1M
  transpose_to_bf16<<<wElem / 256, 256, 0, stream>>>(Wq, WqT, DMODEL, HDIM);
  transpose_to_bf16<<<wElem / 256, 256, 0, stream>>>(Wk, WkT, DMODEL, HDIM);
  transpose_to_bf16<<<wElem / 256, 256, 0, stream>>>(Wv, WvT, DMODEL, HDIM);
  transpose_to_bf16<<<wElem / 256, 256, 0, stream>>>(Wo, WoT, HDIM, DMODEL);

  dim3 gg(HDIM / 64, TB / 128);                 // (16, 32)
  gemm4096x1024_bf16<0><<<gg, 256, 0, stream>>>(qb, WqT, (void*)hq);
  gemm4096x1024_bf16<0><<<gg, 256, 0, stream>>>(kb, WkT, (void*)hk);
  gemm4096x1024_bf16<1><<<gg, 256, 0, stream>>>(vb, WvT, (void*)hvT);

  attn_fused<<<dim3(T_SEQ / 16, BATCH * NHEAD), 32, 0, stream>>>(hq, hk, hvT, avec);

  gemm4096x1024_bf16<2><<<gg, 256, 0, stream>>>(avec, WoT, (void*)attn_out);

  layernorm_rows<<<TB, 256, 0, stream>>>(attn_out, gamma, beta, (float*)d_out);
}
